// MoELayer_72791105733016
// MI455X (gfx1250) — matrix operand order repaired and checked by reading, not compiled, not run on hardware
//
#include <hip/hip_runtime.h>
#include <hip/hip_bf16.h>

#define B_TOK 8192
#define D_DIM 1024
#define E_NUM 8
#define H_DIM 2048

typedef __attribute__((ext_vector_type(16))) __bf16 v16bf;
typedef __attribute__((ext_vector_type(8)))  float  v8f;

union ABRegs { v16bf v; uint4 q[2]; };
struct APtrs { const char* p[4]; };

// -------------------- WMMA 64x64 per-wave GEMM core --------------------
// Per 32-K step, per lane:
//   A tile t: rows slots[t*16 + lane&15]; two b128 at +0/+32 (hi lanes base+16).
//   B tile n: column (colBase + lane&15) of transposed weights; two
//             b128 at +0/+32 (hi lanes base+16).
// Double-buffered (a0/b0 vs a1/b1) so loads for step k+1 are in flight while
// the 16 WMMAs of step k execute -> partial s_wait_loadcnt instead of wait 0.
template <int KSTEPS, int BTS>
__device__ __forceinline__ void wmma_gemm64x64(const APtrs ap, const char* bBase,
                                               v8f acc[16]) {
  ABRegs a0[4], b0[4], a1[4], b1[4];

  auto loadS = [&](ABRegs a[4], ABRegs b[4], int k) {
#pragma unroll
    for (int t = 0; t < 4; ++t) {
      a[t].q[0] = *(const uint4*)(ap.p[t] + 64 * k);
      a[t].q[1] = *(const uint4*)(ap.p[t] + 64 * k + 32);
    }
#pragma unroll
    for (int n = 0; n < 4; ++n) {
      b[n].q[0] = *(const uint4*)(bBase + n * BTS + 64 * k);
      b[n].q[1] = *(const uint4*)(bBase + n * BTS + 64 * k + 32);
    }
  };
  auto comp = [&](ABRegs a[4], ABRegs b[4]) {
#pragma unroll
    for (int n = 0; n < 4; ++n)
#pragma unroll
      for (int t = 0; t < 4; ++t)
        acc[n * 4 + t] = __builtin_amdgcn_wmma_f32_16x16x32_bf16(
            false, a[t].v, false, b[n].v, (short)0, acc[n * 4 + t], false,
            false);
  };

  loadS(a0, b0, 0);
#pragma clang loop unroll(disable)
  for (int k = 0; k < KSTEPS; k += 2) {
    loadS(a1, b1, k + 1);
    comp(a0, b0);
    if (k + 2 < KSTEPS) loadS(a0, b0, k + 2);
    comp(a1, b1);
  }
}

// -------------------- fp32 -> bf16 convert (x) --------------------
__global__ __launch_bounds__(256) void cvt_x_kernel(const float* __restrict__ x,
                                                    __hip_bfloat16* __restrict__ xbf,
                                                    int n) {
  int i = (blockIdx.x * 256 + threadIdx.x) * 4;
  if (i < n) {
    float4 f = *(const float4*)(x + i);
    xbf[i + 0] = __float2bfloat16(f.x);
    xbf[i + 1] = __float2bfloat16(f.y);
    xbf[i + 2] = __float2bfloat16(f.z);
    xbf[i + 3] = __float2bfloat16(f.w);
  }
}

// ---------- transpose + convert: in f32 [Z][R][C] -> out bf16 [Z][C][R] ----------
__global__ __launch_bounds__(256) void tcvt_kernel(const float* __restrict__ in,
                                                   __hip_bfloat16* __restrict__ out,
                                                   int R, int C) {
  __shared__ float tile[32][33];
  const size_t base = (size_t)blockIdx.z * R * C;
  const int c0 = blockIdx.x * 32, r0 = blockIdx.y * 32;
  const int tx = threadIdx.x, ty = threadIdx.y;  // (32, 8)
#pragma unroll
  for (int j = 0; j < 4; ++j)
    tile[ty + j * 8][tx] = in[base + (size_t)(r0 + ty + j * 8) * C + (c0 + tx)];
  __syncthreads();
#pragma unroll
  for (int j = 0; j < 4; ++j)
    out[base + (size_t)(c0 + ty + j * 8) * R + (r0 + tx)] =
        __float2bfloat16(tile[tx][ty + j * 8]);
}

// -------------------- gating: one wave per token --------------------
__global__ __launch_bounds__(256) void gate_kernel(const float* __restrict__ x,
                                                   const float* __restrict__ Wg,
                                                   int* __restrict__ idxList,
                                                   float* __restrict__ wBuf,
                                                   int* __restrict__ counts) {
  const int lane = threadIdx.x & 31;
  const int t = blockIdx.x * 8 + (threadIdx.x >> 5);
  float acc[E_NUM];
#pragma unroll
  for (int e = 0; e < E_NUM; ++e) acc[e] = 0.f;
  const float* xr = x + (size_t)t * D_DIM;
  for (int d = lane; d < D_DIM; d += 32) {
    const float xv = xr[d];
    const float* wr = Wg + (size_t)d * E_NUM;
#pragma unroll
    for (int e = 0; e < E_NUM; ++e) acc[e] += xv * wr[e];
  }
#pragma unroll
  for (int e = 0; e < E_NUM; ++e) {
#pragma unroll
    for (int off = 16; off > 0; off >>= 1)
      acc[e] += __shfl_xor(acc[e], off, 32);
  }
  if (lane == 0) {
    int i0 = 0; float v0 = acc[0];
#pragma unroll
    for (int e = 1; e < E_NUM; ++e)
      if (acc[e] > v0) { v0 = acc[e]; i0 = e; }
    int i1 = -1; float v1 = -3.4e38f;
#pragma unroll
    for (int e = 0; e < E_NUM; ++e)
      if (e != i0 && acc[e] > v1) { v1 = acc[e]; i1 = e; }
    const float e1w = expf(v1 - v0);  // v0 is the max
    const float s = 1.f + e1w;
    const float w0 = 1.f / s, w1 = e1w / s;
    int p0 = atomicAdd(&counts[i0], 1);
    idxList[i0 * B_TOK + p0] = t * 2;
    wBuf[t * 2] = w0;
    int p1 = atomicAdd(&counts[i1], 1);
    idxList[i1 * B_TOK + p1] = t * 2 + 1;
    wBuf[t * 2 + 1] = w1;
  }
}

// ---------- FFN layer 1: h[slot,:] = relu(x[tok] @ W1[e] + b1[e]) ----------
// WG: 64 list rows x 512 cols; 8 waves each 64x64. 4 WGs tile H=2048.
__global__ __launch_bounds__(256) void ffn1_kernel(
    const __hip_bfloat16* __restrict__ xbf, const __hip_bfloat16* __restrict__ w1t,
    const float* __restrict__ b1, const int* __restrict__ idxList,
    const int* __restrict__ counts, __hip_bfloat16* __restrict__ hbuf) {
  const int e = blockIdx.x >> 9;           // 512 blocks per expert
  const int mt = (blockIdx.x >> 2) & 127;  // 128 row tiles of 64
  const int nt = blockIdx.x & 3;           // 4 col tiles of 512
  const int m0 = mt << 6;
  const int cnt = counts[e];
  if (m0 >= cnt) return;

  __shared__ int slots[64];
  if (threadIdx.x < 64) {
    int p = m0 + (int)threadIdx.x;
    if (p > cnt - 1) p = cnt - 1;  // duplicate last row for padding
    slots[threadIdx.x] = idxList[e * B_TOK + p];
  }
  __syncthreads();

  const int lane = threadIdx.x & 31;
  const int w = threadIdx.x >> 5;
  const int hi = lane >> 4;
  const int lid = lane & 15;
  const int colBase = nt * 512 + w * 64;

  APtrs ap;
#pragma unroll
  for (int t = 0; t < 4; ++t)
    ap.p[t] = (const char*)xbf +
              (size_t)(slots[t * 16 + lid] >> 1) * (D_DIM * 2) + hi * 16;
  const char* bBase = (const char*)w1t +
                      (size_t)e * ((size_t)H_DIM * D_DIM * 2) +
                      (size_t)(colBase + lid) * (D_DIM * 2) + hi * 16;

  v8f acc[16];
#pragma unroll
  for (int n = 0; n < 16; ++n) acc[n] = {};
  wmma_gemm64x64<D_DIM / 32, 16 * D_DIM * 2>(ap, bBase, acc);

#pragma unroll
  for (int n = 0; n < 4; ++n) {
    const int col = colBase + n * 16 + lid;
    const float bias = b1[e * H_DIM + col];
#pragma unroll
    for (int t = 0; t < 4; ++t) {
#pragma unroll
      for (int v = 0; v < 8; ++v) {
        const int row = t * 16 + v + hi * 8;
        if (m0 + row < cnt) {
          float val = acc[n * 4 + t][v] + bias;
          val = val > 0.f ? val : 0.f;
          *(__hip_bfloat16*)((char*)hbuf + (size_t)slots[row] * (H_DIM * 2) +
                             col * 2) = __float2bfloat16(val);
        }
      }
    }
  }
}

// ---------- FFN layer 2: out[tok,:] += gate * (h[slot] @ W2[e] + b2[e]) ----------
// WG: 64 list rows x 512 cols; 8 waves each 64x64. 2 WGs tile D=1024.
__global__ __launch_bounds__(256) void ffn2_kernel(
    const __hip_bfloat16* __restrict__ hbuf, const __hip_bfloat16* __restrict__ w2t,
    const float* __restrict__ b2, const int* __restrict__ idxList,
    const int* __restrict__ counts, const float* __restrict__ wBuf,
    float* __restrict__ out) {
  const int e = blockIdx.x >> 8;           // 256 blocks per expert
  const int mt = (blockIdx.x >> 1) & 127;  // 128 row tiles of 64
  const int nt = blockIdx.x & 1;           // 2 col tiles of 512
  const int m0 = mt << 6;
  const int cnt = counts[e];
  if (m0 >= cnt) return;

  __shared__ int slots[64];
  __shared__ float gw[64];
  if (threadIdx.x < 64) {
    int p = m0 + (int)threadIdx.x;
    if (p > cnt - 1) p = cnt - 1;
    const int s = idxList[e * B_TOK + p];
    slots[threadIdx.x] = s;
    gw[threadIdx.x] = wBuf[s];
  }
  __syncthreads();

  const int lane = threadIdx.x & 31;
  const int w = threadIdx.x >> 5;
  const int hi = lane >> 4;
  const int lid = lane & 15;
  const int colBase = nt * 512 + w * 64;

  APtrs ap;
#pragma unroll
  for (int t = 0; t < 4; ++t)
    ap.p[t] = (const char*)hbuf +
              (size_t)slots[t * 16 + lid] * (H_DIM * 2) + hi * 16;
  const char* bBase = (const char*)w2t +
                      (size_t)e * ((size_t)D_DIM * H_DIM * 2) +
                      (size_t)(colBase + lid) * (H_DIM * 2) + hi * 16;

  v8f acc[16];
#pragma unroll
  for (int n = 0; n < 16; ++n) acc[n] = {};
  wmma_gemm64x64<H_DIM / 32, 16 * H_DIM * 2>(ap, bBase, acc);

#pragma unroll
  for (int n = 0; n < 4; ++n) {
    const int col = colBase + n * 16 + lid;
    const float bias = b2[e * D_DIM + col];
#pragma unroll
    for (int t = 0; t < 4; ++t) {
#pragma unroll
      for (int v = 0; v < 8; ++v) {
        const int row = t * 16 + v + hi * 8;
        if (m0 + row < cnt) {
          const float val = (acc[n * 4 + t][v] + bias) * gw[row];
          atomicAdd(&out[(size_t)(slots[row] >> 1) * D_DIM + col], val);
        }
      }
    }
  }
}

// -------------------- host launcher --------------------
extern "C" void kernel_launch(void* const* d_in, const int* in_sizes, int n_in,
                              void* d_out, int out_size, void* d_ws, size_t ws_size,
                              hipStream_t stream) {
  const float* x = (const float*)d_in[0];
  const float* Wg = (const float*)d_in[1];
  const float* W1 = (const float*)d_in[2];
  const float* b1 = (const float*)d_in[3];
  const float* W2 = (const float*)d_in[4];
  const float* b2 = (const float*)d_in[5];
  float* out = (float*)d_out;

  char* ws = (char*)d_ws;
  size_t off = 0;
  auto alloc = [&](size_t bytes) {
    char* p = ws + off;
    off += (bytes + 255) & ~(size_t)255;
    return p;
  };
  __hip_bfloat16* xbf = (__hip_bfloat16*)alloc((size_t)B_TOK * D_DIM * 2);
  __hip_bfloat16* w1t = (__hip_bfloat16*)alloc((size_t)E_NUM * D_DIM * H_DIM * 2);
  __hip_bfloat16* w2t = (__hip_bfloat16*)alloc((size_t)E_NUM * D_DIM * H_DIM * 2);
  __hip_bfloat16* hbuf = (__hip_bfloat16*)alloc((size_t)2 * B_TOK * H_DIM * 2);
  int* idxList = (int*)alloc((size_t)E_NUM * B_TOK * 4);
  float* wBuf = (float*)alloc((size_t)2 * B_TOK * 4);
  int* counts = (int*)alloc(E_NUM * 4);

  hipMemsetAsync(counts, 0, E_NUM * 4, stream);
  hipMemsetAsync(out, 0, (size_t)B_TOK * D_DIM * 4, stream);

  cvt_x_kernel<<<(B_TOK * D_DIM / 4 + 255) / 256, 256, 0, stream>>>(
      x, xbf, B_TOK * D_DIM);
  // W1 [E][D][H] -> w1t [E][H][D]
  tcvt_kernel<<<dim3(H_DIM / 32, D_DIM / 32, E_NUM), dim3(32, 8), 0, stream>>>(
      W1, w1t, D_DIM, H_DIM);
  // W2 [E][H][D] -> w2t [E][D][H]
  tcvt_kernel<<<dim3(D_DIM / 32, H_DIM / 32, E_NUM), dim3(32, 8), 0, stream>>>(
      W2, w2t, H_DIM, D_DIM);
  gate_kernel<<<B_TOK / 8, 256, 0, stream>>>(x, Wg, idxList, wBuf, counts);
  ffn1_kernel<<<E_NUM * 512, 256, 0, stream>>>(xbf, w1t, b1, idxList, counts,
                                               hbuf);
  ffn2_kernel<<<E_NUM * 256, 256, 0, stream>>>(hbuf, w2t, b2, idxList, counts,
                                               wBuf, out);
}
